// RNNDecoder_17892833755475
// MI455X (gfx1250) — matrix operand order repaired and checked by reading, not compiled, not run on hardware
//
#include <hip/hip_runtime.h>
#include <math.h>

// ---------------- fixed problem sizes (from reference) ----------------
#define BATCH 64
#define ISZ   256
#define HSZ   512
#define OSZ   8000
#define TLEN  256
#define XSZ   (ISZ + HSZ)   // 768 = concat([input, z])
#define G3    (3 * HSZ)     // 1536 gate width

#define AS1 __attribute__((address_space(1)))
#define AS3 __attribute__((address_space(3)))

typedef __attribute__((ext_vector_type(4))) int v4i;

// gfx1250 async global->LDS copy builtins (toolchain-dependent; bridge doc)
#if defined(__has_builtin)
#  if __has_builtin(__builtin_amdgcn_global_load_async_to_lds_b128)
#    define HAVE_ASYNC_LDS 1
#  endif
#  if __has_builtin(__builtin_amdgcn_s_wait_asynccnt)
#    define HAVE_WAIT_ASYNC 1
#  endif
#endif
#ifndef HAVE_ASYNC_LDS
#  define HAVE_ASYNC_LDS 0
#endif
#ifndef HAVE_WAIT_ASYNC
#  define HAVE_WAIT_ASYNC 0
#endif

__device__ __forceinline__ void wait_async0() {
#if HAVE_WAIT_ASYNC
  __builtin_amdgcn_s_wait_asynccnt(0);
#else
  asm volatile("s_wait_asynccnt 0" ::: "memory");
#endif
}

// ---------------- WMMA types ----------------
typedef __attribute__((ext_vector_type(16))) __bf16 v16bf;
typedef __attribute__((ext_vector_type(8)))  __bf16 v8bf;
typedef __attribute__((ext_vector_type(8)))  float  v8f;

union ABf { v16bf v; v8bf h[2]; };

__device__ __forceinline__ __bf16 f2bf(float f) {
  unsigned u = __builtin_bit_cast(unsigned, f);
  unsigned r = u + 0x7FFFu + ((u >> 16) & 1u);   // round-to-nearest-even
  unsigned short hv = (unsigned short)(r >> 16);
  return __builtin_bit_cast(__bf16, hv);
}

__device__ __forceinline__ v8f zero8() {
  v8f v;
#pragma unroll
  for (int i = 0; i < 8; ++i) v[i] = 0.0f;
  return v;
}

__device__ __forceinline__ v8f wmma_bf16(v16bf a, v16bf b, v8f c) {
  // D = A(16x32 bf16) * B(32x16 bf16) + C(16x16 f32)
  return __builtin_amdgcn_wmma_f32_16x16x32_bf16(false, a, false, b,
                                                 (short)0, c, false, false);
}

// Load A fragment: lane L -> row M = L%16, K = (L/16)*16 + j
// => two contiguous 8-elem (16B) loads per lane.
__device__ __forceinline__ v16bf load_a(const __bf16* row, int ahalf) {
  ABf a;
  a.h[0] = *(const v8bf*)(row + 2 * ahalf);
  a.h[1] = *(const v8bf*)(row + 2 * ahalf + 8);
  return a.v;
}

// ---------------- weight tiling: fp32 row-major [N,K] -> bf16 WMMA-B tiles --
// Tile (nt,kt) holds 512 bf16: element L*16+j = W[nt*16 + L%16][kt*32 + (L/16)*16 + j]
__global__ void tile_weights(const float* __restrict__ src, __bf16* __restrict__ dst,
                             int K, int KT, long total) {
  long e = (long)blockIdx.x * blockDim.x + threadIdx.x;
  if (e >= total) return;
  int  r    = (int)(e & 511);
  long tile = e >> 9;
  int  L = r >> 4, j = r & 15;
  int  nt = (int)(tile / KT), kt = (int)(tile % KT);
  int  n = nt * 16 + (L & 15);
  int  k = kt * 32 + (L >> 4) * 16 + j;
  dst[e] = f2bf(src[(long)n * K + k]);
}

// ---------------- initial state: x0 = [ones, z] (bf16), h0 = z --------------
__global__ void init_state(const float* __restrict__ z,
                           __bf16* __restrict__ xbf,
                           float* __restrict__ hf,
                           __bf16* __restrict__ hbf) {
  int idx = blockIdx.x * blockDim.x + threadIdx.x;   // over BATCH*XSZ
  if (idx >= BATCH * XSZ) return;
  int b = idx / XSZ, c = idx % XSZ;
  xbf[idx] = (c < ISZ) ? f2bf(1.0f) : f2bf(z[b * HSZ + (c - ISZ)]);
  if (idx < BATCH * HSZ) {
    float v = z[idx];
    hf[idx]  = v;
    hbf[idx] = f2bf(v);
  }
}

// ---------------- per-step GEMM: gi = x@Wih^T + b_ih, gh = h@Whh^T + b_hh ----
// 384 waves, one 16x16 output tile each (4 M-tiles x 96 N-tiles).
__global__ void __launch_bounds__(256) step_gates_gemm(
    const __bf16* __restrict__ xbf,   // [64 x 768] row-major bf16
    const __bf16* __restrict__ hbf,   // [64 x 512] row-major bf16
    const __bf16* __restrict__ WihT,  // tiled B, 96 x 24 tiles
    const __bf16* __restrict__ WhhT,  // tiled B, 96 x 16 tiles
    const float*  __restrict__ b_ih,
    const float*  __restrict__ b_hh,
    float* __restrict__ gi, float* __restrict__ gh) {
  int wave = (blockIdx.x * blockDim.x + threadIdx.x) >> 5;
  int lane = threadIdx.x & 31;
  int mt = wave & 3;          // 64/16 = 4 M tiles
  int nt = wave >> 2;         // 1536/16 = 96 N tiles
  int mrow  = mt * 16 + (lane & 15);
  int ahalf = (lane >> 4) * 8;

  v8f acc_i = zero8();
  v8f acc_h = zero8();

  const __bf16* xr = xbf + (long)mrow * XSZ;
#pragma unroll 4
  for (int kt = 0; kt < XSZ / 32; ++kt) {       // K = 768
    v16bf a = load_a(xr + kt * 32, ahalf);
    v16bf b = *(const v16bf*)(WihT + ((long)(nt * (XSZ / 32) + kt) << 9) + lane * 16);
    acc_i = wmma_bf16(a, b, acc_i);
  }
  const __bf16* hr = hbf + (long)mrow * HSZ;
#pragma unroll 4
  for (int kt = 0; kt < HSZ / 32; ++kt) {       // K = 512
    v16bf a = load_a(hr + kt * 32, ahalf);
    v16bf b = *(const v16bf*)(WhhT + ((long)(nt * (HSZ / 32) + kt) << 9) + lane * 16);
    acc_h = wmma_bf16(a, b, acc_h);
  }

  int n  = nt * 16 + (lane & 15);
  float bi = b_ih[n], bh = b_hh[n];
  int mbase = mt * 16 + (lane >> 4) * 8;        // C layout: M = (lane/16)*8 + r
#pragma unroll
  for (int r = 0; r < 8; ++r) {
    gi[(long)(mbase + r) * G3 + n] = acc_i[r] + bi;
    gh[(long)(mbase + r) * G3 + n] = acc_h[r] + bh;
  }
}

// ---------------- GRU gate pointwise + history append -----------------------
__global__ void gru_pointwise(const float* __restrict__ gi,
                              const float* __restrict__ gh,
                              float* __restrict__ hf,
                              __bf16* __restrict__ hbf,
                              __bf16* __restrict__ hs, int t) {
  int idx = blockIdx.x * blockDim.x + threadIdx.x;   // over BATCH*HSZ
  if (idx >= BATCH * HSZ) return;
  int b = idx >> 9, j = idx & (HSZ - 1);
  const float* gir = gi + (long)b * G3;
  const float* ghr = gh + (long)b * G3;
  float r  = 1.0f / (1.0f + __expf(-(gir[j]           + ghr[j])));
  float u  = 1.0f / (1.0f + __expf(-(gir[HSZ + j]     + ghr[HSZ + j])));
  float nn = tanhf(gir[2 * HSZ + j] + r * ghr[2 * HSZ + j]);
  float hnew = (1.0f - u) * nn + u * hf[idx];
  hf[idx] = hnew;
  __bf16 hb = f2bf(hnew);
  hbf[idx] = hb;
  hs[(long)t * BATCH * HSZ + idx] = hb;   // row t*64 + b of [T*B, H]
}

// ---------------- per-step h2x: x_next = relu(h@Wh2x^T + b) -----------------
// 64 waves (4 M-tiles x 16 N-tiles), writes bf16 into first 256 cols of x.
__global__ void __launch_bounds__(256) h2x_gemm(
    const __bf16* __restrict__ hbf,    // [64 x 512]
    const __bf16* __restrict__ Wt,     // tiled B, 16 x 16 tiles
    const float*  __restrict__ bias,
    __bf16* __restrict__ xbf) {        // [64 x 768], cols 0..255
  int wave = (blockIdx.x * blockDim.x + threadIdx.x) >> 5;
  int lane = threadIdx.x & 31;
  int mt = wave & 3;
  int nt = wave >> 2;                  // 256/16 = 16
  int mrow  = mt * 16 + (lane & 15);
  int ahalf = (lane >> 4) * 8;

  v8f acc = zero8();
  const __bf16* hr = hbf + (long)mrow * HSZ;
#pragma unroll 4
  for (int kt = 0; kt < HSZ / 32; ++kt) {
    v16bf a = load_a(hr + kt * 32, ahalf);
    v16bf b = *(const v16bf*)(Wt + ((long)(nt * (HSZ / 32) + kt) << 9) + lane * 16);
    acc = wmma_bf16(a, b, acc);
  }

  int n  = nt * 16 + (lane & 15);
  float bv = bias[n];
  int mbase = mt * 16 + (lane >> 4) * 8;
#pragma unroll
  for (int r = 0; r < 8; ++r) {
    float v = acc[r] + bv;
    if (v < 0.0f) v = 0.0f;
    xbf[(long)(mbase + r) * XSZ + n] = f2bf(v);
  }
}

// ---------------- final GEMM: out = hs @ Wout^T + b_out ---------------------
// Block = 8 waves = 8 M-tiles x 4 N-tiles (128x64 f32 output super-tile).
// The 4 consecutive N-tiles form a CONTIGUOUS 64KB slab of tiled W_out, which
// is staged once into LDS (async DMA when available) and shared by all waves.
// Grid: 128 M-blocks x 125 N-groups = 16000 blocks.
__global__ void __launch_bounds__(256) out_gemm(
    const __bf16* __restrict__ hs,     // [16384 x 512] bf16 row-major
    const __bf16* __restrict__ Wt,     // tiled B, 500 x 16 tiles
    const float*  __restrict__ bias,
    float* __restrict__ out) {
  __shared__ __align__(16) __bf16 Bs[4 * 16 * 512];   // 64 KB B panel

  int wave = (int)(threadIdx.x >> 5);
  int lane = (int)(threadIdx.x & 31);
  int mblock = (int)(blockIdx.x & 127);   // 1024/8 = 128 M-blocks
  int ng     = (int)(blockIdx.x >> 7);    // 125 groups of 4 N-tiles
  int n0t    = ng * 4;

  int mt    = mblock * 8 + wave;
  int mrow  = mt * 16 + (lane & 15);
  int ahalf = (lane >> 4) * 8;
  const __bf16* ar = hs + (long)mrow * HSZ;
  __builtin_prefetch(ar, 0, 0);           // global_prefetch_b8: warm A row

  // ---- stage 64KB contiguous B panel: Wt[n0t*16 .. (n0t+4)*16) tiles ----
  const __bf16* gB = Wt + ((long)(n0t * 16) << 9);
#if HAVE_ASYNC_LDS
#pragma unroll
  for (int it = 0; it < 16; ++it) {
    int eoff = (it * 256 + (int)threadIdx.x) * 8;   // 16B per thread per iter
    __builtin_amdgcn_global_load_async_to_lds_b128(
        (AS1 v4i*)(gB + eoff), (AS3 v4i*)(Bs + eoff), 0, 0);
  }
  wait_async0();
#else
#pragma unroll
  for (int it = 0; it < 16; ++it) {
    int eoff = (it * 256 + (int)threadIdx.x) * 8;
    *(v8bf*)(Bs + eoff) = *(const v8bf*)(gB + eoff);
  }
#endif
  __syncthreads();

  // ---- compute: A from global, B from LDS (shared by 8 waves) ----
  v8f acc[4];
#pragma unroll
  for (int q = 0; q < 4; ++q) acc[q] = zero8();

  for (int kt = 0; kt < HSZ / 32; ++kt) {
    v16bf a = load_a(ar + kt * 32, ahalf);
#pragma unroll
    for (int q = 0; q < 4; ++q) {
      v16bf b = *(const v16bf*)(Bs + (((q * 16 + kt) << 9) + lane * 16));
      acc[q] = wmma_bf16(a, b, acc[q]);
    }
  }

  int mbase = mt * 16 + (lane >> 4) * 8;
#pragma unroll
  for (int q = 0; q < 4; ++q) {
    int n = (n0t + q) * 16 + (lane & 15);
    float bv = bias[n];
#pragma unroll
    for (int r = 0; r < 8; ++r)
      out[(long)(mbase + r) * OSZ + n] = acc[q][r] + bv;
  }
}

// ---------------- host side -------------------------------------------------
extern "C" void kernel_launch(void* const* d_in, const int* in_sizes, int n_in,
                              void* d_out, int out_size, void* d_ws, size_t ws_size,
                              hipStream_t stream) {
  (void)in_sizes; (void)n_in; (void)out_size; (void)ws_size;
  const float* z      = (const float*)d_in[0];
  // d_in[1] = input_length (fixed 256 in reference setup)
  const float* W_ih   = (const float*)d_in[2];
  const float* b_ih   = (const float*)d_in[3];
  const float* W_hh   = (const float*)d_in[4];
  const float* b_hh   = (const float*)d_in[5];
  const float* W_h2x  = (const float*)d_in[6];
  const float* b_h2x  = (const float*)d_in[7];
  const float* W_out  = (const float*)d_in[8];
  const float* b_out  = (const float*)d_in[9];
  float* out = (float*)d_out;

  // workspace carve (all regions fully re-written every launch)
  size_t off = 0;
  auto carve = [&](size_t bytes) -> char* {
    char* p = (char*)d_ws + off;
    off += (bytes + 255) & ~(size_t)255;
    return p;
  };
  __bf16* WihT  = (__bf16*)carve((size_t)96  * 24 * 512 * 2);
  __bf16* WhhT  = (__bf16*)carve((size_t)96  * 16 * 512 * 2);
  __bf16* Wh2xT = (__bf16*)carve((size_t)16  * 16 * 512 * 2);
  __bf16* WoutT = (__bf16*)carve((size_t)500 * 16 * 512 * 2);
  __bf16* xbf   = (__bf16*)carve((size_t)BATCH * XSZ * 2);
  float*  hf    = (float*) carve((size_t)BATCH * HSZ * 4);
  __bf16* hbf   = (__bf16*)carve((size_t)BATCH * HSZ * 2);
  float*  gi    = (float*) carve((size_t)BATCH * G3  * 4);
  float*  gh    = (float*) carve((size_t)BATCH * G3  * 4);
  __bf16* hs    = (__bf16*)carve((size_t)TLEN * BATCH * HSZ * 2);

  // 1) weight tiling (fp32 -> bf16 WMMA-B layout)
  {
    long t1 = (long)96 * 24 * 512;
    tile_weights<<<(int)((t1 + 255) / 256), 256, 0, stream>>>(W_ih,  WihT,  XSZ, 24, t1);
    long t2 = (long)96 * 16 * 512;
    tile_weights<<<(int)((t2 + 255) / 256), 256, 0, stream>>>(W_hh,  WhhT,  HSZ, 16, t2);
    long t3 = (long)16 * 16 * 512;
    tile_weights<<<(int)((t3 + 255) / 256), 256, 0, stream>>>(W_h2x, Wh2xT, HSZ, 16, t3);
    long t4 = (long)500 * 16 * 512;
    tile_weights<<<(int)((t4 + 255) / 256), 256, 0, stream>>>(W_out, WoutT, HSZ, 16, t4);
  }

  // 2) initial state
  init_state<<<(BATCH * XSZ + 255) / 256, 256, 0, stream>>>(z, xbf, hf, hbf);

  // 3) sequential GRU recurrence (stream order enforces dependences)
  for (int t = 0; t < TLEN; ++t) {
    step_gates_gemm<<<48, 256, 0, stream>>>(xbf, hbf, WihT, WhhT, b_ih, b_hh, gi, gh);
    gru_pointwise<<<(BATCH * HSZ + 255) / 256, 256, 0, stream>>>(gi, gh, hf, hbf, hs, t);
    h2x_gemm<<<8, 256, 0, stream>>>(hbf, Wh2xT, b_h2x, xbf);
  }

  // 4) output projection: [16384 x 512] @ [512 x 8000] + bias
  out_gemm<<<16000, 256, 0, stream>>>(hs, WoutT, b_out, out);
}
